// diffusionNetworks_58755152609567
// MI455X (gfx1250) — matrix operand order repaired and checked by reading, not compiled, not run on hardware
//
#include <hip/hip_runtime.h>
#include <hip/hip_bf16.h>

// CDNA5 (gfx1250, wave32) graph diffusion network.
// All 64x64 matmuls on v_wmma_f32_16x16x32_bf16 (bf16 in, f32 accum).
// Node state is node-major (N,64): one node = one contiguous 256B run.
// Edge kernels are persistent (grid-stride over 128-edge tiles) so LDS weight
// staging amortizes; tv_norm stats accumulate in VGPRs, flushed once.

#define HSTEP 0.1f
#define EPSV  1e-3f

typedef __attribute__((ext_vector_type(16))) __bf16 v16bf;
typedef __attribute__((ext_vector_type(8)))  float  v8f;

#define V8F_ZERO {0.f,0.f,0.f,0.f,0.f,0.f,0.f,0.f}

__device__ __forceinline__ v8f wmma_bf16(v16bf a, v16bf b, v8f c) {
  // 8 args: (neg_a, A, neg_b, B, c_mod, C, reuse_a, reuse_b)
  return __builtin_amdgcn_wmma_f32_16x16x32_bf16(false, a, false, b, (short)0, c,
                                                 false, false);
}

// A-fragment (16x32 bf16) K-offset for packed pair v (0..7), lane half hi.
// Contiguous K (matches B): lanes 0-15 hold K=0..15, lanes 16-31 K=16..31.
__device__ __forceinline__ int a_koff(int v, int hi) {
  return (v << 1) + (hi << 4);
}

// Build A fragment for row-block r (0..3), K-block k (0..1) from a 64x64
// row-major bf16 matrix staged in LDS (lowers to ds_load_b128 pairs).
__device__ __forceinline__ v16bf load_a_frag(const __bf16* sW, int r, int k,
                                             int ln, int hi) {
  v16bf a;
  const __bf16* row = sW + (r * 16 + ln) * 64 + k * 32;
#pragma unroll
  for (int v = 0; v < 8; ++v) {
    int kk = a_koff(v, hi);
    a[2 * v]     = row[kk];
    a[2 * v + 1] = row[kk + 1];
  }
  return a;
}

// B fragment (32x16 bf16) from a node-major row (contiguous 64 floats):
// element e of b[k] is K = k*32 + hi*16 + e  ->  4 float4 loads per k-block.
__device__ __forceinline__ v16bf load_b_nc(const float* __restrict__ Xrow,
                                           int k, int hi) {
  const float4* p = (const float4*)(Xrow + k * 32 + hi * 16);
  v16bf b;
#pragma unroll
  for (int q = 0; q < 4; ++q) {
    float4 f = p[q];
    b[4 * q + 0] = (__bf16)f.x;
    b[4 * q + 1] = (__bf16)f.y;
    b[4 * q + 2] = (__bf16)f.z;
    b[4 * q + 3] = (__bf16)f.w;
  }
  return b;
}

// Edge-gradient B fragment: g = x[i,:] - x[j,:] (both rows contiguous).
__device__ __forceinline__ void edge_grad_b_nc(const float* __restrict__ Xi,
                                               const float* __restrict__ Xj,
                                               int hi, v16bf b[2]) {
#pragma unroll
  for (int k = 0; k < 2; ++k) {
    const float4* pi = (const float4*)(Xi + k * 32 + hi * 16);
    const float4* pj = (const float4*)(Xj + k * 32 + hi * 16);
#pragma unroll
    for (int q = 0; q < 4; ++q) {
      float4 a = pi[q], c = pj[q];
      b[k][4 * q + 0] = (__bf16)(a.x - c.x);
      b[k][4 * q + 1] = (__bf16)(a.y - c.y);
      b[k][4 * q + 2] = (__bf16)(a.z - c.z);
      b[k][4 * q + 3] = (__bf16)(a.w - c.w);
    }
  }
}

// ---------------------------------------------------------------------------
// Dense node GEMM: Y = op(W) @ X.  IN_NC/OUT_NC select (N,64) node-major vs
// (64,N) channel-major layouts.  One wave = 16 columns, block = 128 columns.
// ---------------------------------------------------------------------------
template <int IN_NC, int OUT_NC>
__global__ __launch_bounds__(256)
void gemm64_kernel(const float* __restrict__ W, const float* __restrict__ X,
                   float* __restrict__ Y, int N, int transW) {
  __shared__ __bf16 sW[64 * 64];
  for (int t = threadIdx.x; t < 4096; t += 256) {
    int rr = t >> 6, cc = t & 63;
    sW[t] = (__bf16)(transW ? W[cc * 64 + rr] : W[t]);
  }
  __syncthreads();

  int wave = threadIdx.x >> 5, lane = threadIdx.x & 31;
  int hi = lane >> 4, ln = lane & 15;
  int n  = blockIdx.x * 128 + wave * 16 + ln;
  int nc = n < N ? n : N - 1;  // clamp: all lanes stay active (EXEC all-1s)

  v16bf b[2];
  if (IN_NC) {
#pragma unroll
    for (int k = 0; k < 2; ++k) b[k] = load_b_nc(X + (size_t)nc * 64, k, hi);
  } else {
#pragma unroll
    for (int k = 0; k < 2; ++k)
#pragma unroll
      for (int v = 0; v < 8; ++v) {
        int K = k * 32 + v * 2 + hi * 16;
        b[k][2 * v]     = (__bf16)X[(size_t)K * N + nc];
        b[k][2 * v + 1] = (__bf16)X[(size_t)(K + 1) * N + nc];
      }
  }

#pragma unroll
  for (int r = 0; r < 4; ++r) {
    v8f acc = V8F_ZERO;
#pragma unroll
    for (int k = 0; k < 2; ++k)
      acc = wmma_bf16(load_a_frag(sW, r, k, ln, hi), b[k], acc);
    if (n < N) {
      if (OUT_NC) {
        float4 lo = {acc[0], acc[1], acc[2], acc[3]};
        float4 hv = {acc[4], acc[5], acc[6], acc[7]};
        float* dst = Y + (size_t)n * 64 + r * 16 + hi * 8;
        *(float4*)dst       = lo;
        *(float4*)(dst + 4) = hv;
      } else {
#pragma unroll
        for (int v = 0; v < 8; ++v)
          Y[(size_t)(r * 16 + hi * 8 + v) * N + n] = acc[v];
      }
    }
  }
}

// ---------------------------------------------------------------------------
// Edge pass 1 (persistent): per-channel sum / sumsq of Kn@gradX and Ke@gradX.
// Stats accumulate in VGPRs across the tile loop; single LDS+global flush.
// Edge intermediates (2 x 410MB/layer) never touch memory.
// ---------------------------------------------------------------------------
__global__ __launch_bounds__(256)
void edge_stats_kernel(const float* __restrict__ X, const int* __restrict__ iInd,
                       const int* __restrict__ jInd, const float* __restrict__ Kn,
                       const float* __restrict__ Ke, int E, int numTiles,
                       float* __restrict__ stats /* [4][64] */) {
  __shared__ __bf16 sWd[4096];
  __shared__ __bf16 sWe[4096];
  __shared__ float  sred[256];
  for (int t = threadIdx.x; t < 4096; t += 256) {
    sWd[t] = (__bf16)Kn[t];
    sWe[t] = (__bf16)Ke[t];
  }
  sred[threadIdx.x] = 0.f;
  __syncthreads();

  int wave = threadIdx.x >> 5, lane = threadIdx.x & 31;
  int hi = lane >> 4, ln = lane & 15;

  // per-lane register accumulators: 4 stats x 32 channel slots
  float s0[32], s1[32], s2[32], s3[32];
#pragma unroll
  for (int q = 0; q < 32; ++q) { s0[q] = 0.f; s1[q] = 0.f; s2[q] = 0.f; s3[q] = 0.f; }

  for (int tile = blockIdx.x; tile < numTiles; tile += gridDim.x) {
    int e  = tile * 128 + wave * 16 + ln;
    int ec = e < E ? e : E - 1;
    int ii = iInd[ec], jj = jInd[ec];
    float msk = (e < E) ? 1.f : 0.f;

    v16bf b[2];
    edge_grad_b_nc(X + (size_t)ii * 64, X + (size_t)jj * 64, hi, b);

#pragma unroll
    for (int r = 0; r < 4; ++r) {
      v8f ad = V8F_ZERO, aa = V8F_ZERO;
#pragma unroll
      for (int k = 0; k < 2; ++k) {
        ad = wmma_bf16(load_a_frag(sWd, r, k, ln, hi), b[k], ad);
        aa = wmma_bf16(load_a_frag(sWe, r, k, ln, hi), b[k], aa);
      }
#pragma unroll
      for (int v = 0; v < 8; ++v) {
        float vd = ad[v] * msk, va = aa[v] * msk;
        s0[r * 8 + v] += vd;
        s1[r * 8 + v] += vd * vd;
        s2[r * 8 + v] += va;
        s3[r * 8 + v] += va * va;
      }
    }
  }

  // one LDS flush per lane (ds_add_f32), then one global atomic per channel
#pragma unroll
  for (int r = 0; r < 4; ++r)
#pragma unroll
    for (int v = 0; v < 8; ++v) {
      int ch = r * 16 + hi * 8 + v;
      atomicAdd(&sred[ch],        s0[r * 8 + v]);
      atomicAdd(&sred[64 + ch],   s1[r * 8 + v]);
      atomicAdd(&sred[128 + ch],  s2[r * 8 + v]);
      atomicAdd(&sred[192 + ch],  s3[r * 8 + v]);
    }
  __syncthreads();
  atomicAdd(&stats[threadIdx.x], sred[threadIdx.x]);
}

// ---------------------------------------------------------------------------
// Edge pass 2 (persistent): recompute GEMMs, tv_norm + ReLU, scatter into
// div (+/-) and adv (+,+, *0.5) with f32 atomics (accumulators L2-resident).
// fin staged to LDS once per block: [0]=mean_d [64]=scale_d [128]=mean_a [192]=scale_a
// ---------------------------------------------------------------------------
__global__ __launch_bounds__(256)
void edge_apply_kernel(const float* __restrict__ X, const int* __restrict__ iInd,
                       const int* __restrict__ jInd, const float* __restrict__ Kn,
                       const float* __restrict__ Ke, const float* __restrict__ fin,
                       float* __restrict__ bufD, float* __restrict__ bufA,
                       int E, int numTiles) {
  __shared__ __bf16 sWd[4096];
  __shared__ __bf16 sWe[4096];
  __shared__ float  sfin[256];
  for (int t = threadIdx.x; t < 4096; t += 256) {
    sWd[t] = (__bf16)Kn[t];
    sWe[t] = (__bf16)Ke[t];
  }
  sfin[threadIdx.x] = fin[threadIdx.x];
  __syncthreads();

  int wave = threadIdx.x >> 5, lane = threadIdx.x & 31;
  int hi = lane >> 4, ln = lane & 15;

  for (int tile = blockIdx.x; tile < numTiles; tile += gridDim.x) {
    int e  = tile * 128 + wave * 16 + ln;
    int ec = e < E ? e : E - 1;
    int ii = iInd[ec], jj = jInd[ec];

    v16bf b[2];
    edge_grad_b_nc(X + (size_t)ii * 64, X + (size_t)jj * 64, hi, b);

    v8f accd[4], acca[4];
#pragma unroll
    for (int r = 0; r < 4; ++r) {
      accd[r] = (v8f)V8F_ZERO;
      acca[r] = (v8f)V8F_ZERO;
#pragma unroll
      for (int k = 0; k < 2; ++k) {
        accd[r] = wmma_bf16(load_a_frag(sWd, r, k, ln, hi), b[k], accd[r]);
        acca[r] = wmma_bf16(load_a_frag(sWe, r, k, ln, hi), b[k], acca[r]);
      }
    }

    if (e < E) {
      float* di = bufD + (size_t)ii * 64;
      float* dj = bufD + (size_t)jj * 64;
      float* ai = bufA + (size_t)ii * 64;
      float* aj = bufA + (size_t)jj * 64;
#pragma unroll
      for (int r = 0; r < 4; ++r)
#pragma unroll
        for (int v = 0; v < 8; ++v) {
          int ch = r * 16 + hi * 8 + v;
          float vd = fmaxf((accd[r][v] - sfin[ch]) * sfin[64 + ch], 0.f);
          float va = fmaxf((acca[r][v] - sfin[128 + ch]) * sfin[192 + ch], 0.f) * 0.5f;
          atomicAdd(di + ch,  vd);
          atomicAdd(dj + ch, -vd);
          atomicAdd(ai + ch,  va);
          atomicAdd(aj + ch,  va);
        }
    }
  }
}

// ---------------------------------------------------------------------------
// Per-channel sum/sumsq over N for the reaction branch, node-major layout:
// threads (c, g) scan rows n = 4*blk + g  ->  fully coalesced 256B bursts.
// ---------------------------------------------------------------------------
__global__ __launch_bounds__(256)
void rowstats_kernel(const float* __restrict__ X, int N, float* __restrict__ out) {
  int c = threadIdx.x & 63, g = threadIdx.x >> 6;  // 4 row groups
  float s = 0.f, ss = 0.f;
  for (int n = blockIdx.x * 4 + g; n < N; n += gridDim.x * 4) {
    float v = X[(size_t)n * 64 + c];
    s += v; ss += v * v;
  }
  __shared__ float rs[256], rss[256];
  rs[threadIdx.x] = s; rss[threadIdx.x] = ss;
  __syncthreads();
  if (g == 0) {
    s  = rs[c]  + rs[64 + c]  + rs[128 + c]  + rs[192 + c];
    ss = rss[c] + rss[64 + c] + rss[128 + c] + rss[192 + c];
    atomicAdd(&out[c], s);
    atomicAdd(&out[64 + c], ss);
  }
}

// raw: [6][64] (sum_d, sq_d, sum_a, sq_a, sum_r, sq_r) -> fin [6][64]
// tv_norm: sum((X-mean)^2) = sumsq - cnt*mean^2  ->  scale = rsqrt(. + eps)
__global__ void finalize_stats_kernel(const float* __restrict__ raw, float Ecnt,
                                      float Ncnt, float* __restrict__ fin) {
  int c = threadIdx.x;
  if (c < 64) {
    float md = raw[c] / Ecnt;
    fin[c]       = md;
    fin[64 + c]  = rsqrtf(raw[64 + c] - Ecnt * md * md + EPSV);
    float ma = raw[128 + c] / Ecnt;
    fin[128 + c] = ma;
    fin[192 + c] = rsqrtf(raw[192 + c] - Ecnt * ma * ma + EPSV);
    float mr = raw[256 + c] / Ncnt;
    fin[256 + c] = mr;
    fin[320 + c] = rsqrtf(raw[320 + c] - Ncnt * mr * mr + EPSV);
  }
}

// ---------------------------------------------------------------------------
// Node update: Jdiff = Kn^T @ div via WMMA, fused with
// x -= H * (adv + Jdiff + relu(tv_norm(Kr@x)))   (in place, node-major).
// ---------------------------------------------------------------------------
__global__ __launch_bounds__(256)
void node_update_kernel(float* __restrict__ X, const float* __restrict__ bufD,
                        const float* __restrict__ bufA,
                        const float* __restrict__ rtemp,
                        const float* __restrict__ Kn,
                        const float* __restrict__ fin, int N) {
  __shared__ __bf16 sWt[4096];
  __shared__ float  sfr[128];  // mean_r | scale_r
  for (int t = threadIdx.x; t < 4096; t += 256) {
    int rr = t >> 6, cc = t & 63;
    sWt[t] = (__bf16)Kn[cc * 64 + rr];  // Kn^T
  }
  if (threadIdx.x < 128) sfr[threadIdx.x] = fin[256 + threadIdx.x];
  __syncthreads();

  int wave = threadIdx.x >> 5, lane = threadIdx.x & 31;
  int hi = lane >> 4, ln = lane & 15;
  int n  = blockIdx.x * 128 + wave * 16 + ln;
  int nc = n < N ? n : N - 1;

  v16bf b[2];
#pragma unroll
  for (int k = 0; k < 2; ++k) b[k] = load_b_nc(bufD + (size_t)nc * 64, k, hi);

#pragma unroll
  for (int r = 0; r < 4; ++r) {
    v8f acc = V8F_ZERO;
#pragma unroll
    for (int k = 0; k < 2; ++k)
      acc = wmma_bf16(load_a_frag(sWt, r, k, ln, hi), b[k], acc);
    if (n < N) {
      size_t base = (size_t)n * 64 + r * 16 + hi * 8;
      const float4* rt4 = (const float4*)(rtemp + base);
      const float4* av4 = (const float4*)(bufA + base);
      float4* x4 = (float4*)(X + base);
      float4 rt[2] = {rt4[0], rt4[1]};
      float4 av[2] = {av4[0], av4[1]};
      float4 xv[2] = {x4[0], x4[1]};
      float* rtf = (float*)rt;
      float* avf = (float*)av;
      float* xvf = (float*)xv;
#pragma unroll
      for (int v = 0; v < 8; ++v) {
        int ch = r * 16 + hi * 8 + v;
        float ri = fmaxf((rtf[v] - sfr[ch]) * sfr[64 + ch], 0.f);
        xvf[v] -= HSTEP * (avf[v] + acc[v] + ri);
      }
      x4[0] = xv[0];
      x4[1] = xv[1];
    }
  }
}

// ---------------------------------------------------------------------------
extern "C" void kernel_launch(void* const* d_in, const int* in_sizes, int n_in,
                              void* d_out, int out_size, void* d_ws, size_t ws_size,
                              hipStream_t stream) {
  (void)n_in; (void)out_size; (void)ws_size;
  const float* xn      = (const float*)d_in[0];
  const int*   iInd    = (const int*)d_in[1];
  const int*   jInd    = (const int*)d_in[2];
  const float* KNopen  = (const float*)d_in[3];
  const float* KNclose = (const float*)d_in[4];
  const float* KN      = (const float*)d_in[5];
  const float* KE      = (const float*)d_in[6];
  const float* KR      = (const float*)d_in[7];
  float* out = (float*)d_out;

  const int C = 64;
  const int N = in_sizes[0] / C;
  const int E = in_sizes[1];
  const int L = in_sizes[5] / (C * C);
  const size_t cn = (size_t)C * (size_t)N;

  // workspace carve (fp32, node-major): x | rtemp | div | adv | stats | fin
  float* x     = (float*)d_ws;
  float* rtemp = x + cn;
  float* bufD  = rtemp + cn;
  float* bufA  = bufD + cn;
  float* sraw  = bufA + cn;
  float* fin   = sraw + 6 * 64;

  dim3 blk(256);
  int gN = (N + 127) / 128;
  int numTiles = (E + 127) / 128;
  int gE = numTiles < 1024 ? numTiles : 1024;  // persistent edge blocks

  // x = KNopen @ xn   ((C,N) input -> node-major x)
  gemm64_kernel<0, 1><<<gN, blk, 0, stream>>>(KNopen, xn, x, N, 0);

  for (int l = 0; l < L; ++l) {
    const float* Kn = KN + (size_t)l * C * C;
    const float* Ke = KE + (size_t)l * C * C;
    const float* Kr = KR + (size_t)l * C * C;

    hipMemsetAsync(bufD, 0, cn * sizeof(float), stream);
    hipMemsetAsync(bufA, 0, cn * sizeof(float), stream);
    hipMemsetAsync(sraw, 0, 6 * 64 * sizeof(float), stream);

    edge_stats_kernel<<<gE, blk, 0, stream>>>(x, iInd, jInd, Kn, Ke, E,
                                              numTiles, sraw);
    gemm64_kernel<1, 1><<<gN, blk, 0, stream>>>(Kr, x, rtemp, N, 0);
    rowstats_kernel<<<128, blk, 0, stream>>>(rtemp, N, sraw + 4 * 64);
    finalize_stats_kernel<<<1, 64, 0, stream>>>(sraw, (float)E, (float)N, fin);
    edge_apply_kernel<<<gE, blk, 0, stream>>>(x, iInd, jInd, Kn, Ke, fin,
                                              bufD, bufA, E, numTiles);
    node_update_kernel<<<gN, blk, 0, stream>>>(x, bufD, bufA, rtemp, Kn, fin, N);
  }

  // out = KNclose @ x   (node-major x -> (C,N) output)
  gemm64_kernel<1, 0><<<gN, blk, 0, stream>>>(KNclose, x, out, N, 0);
}
